// RowAreaAttention_46815143526901
// MI455X (gfx1250) — matrix operand order repaired and checked by reading, not compiled, not run on hardware
//
#include <hip/hip_runtime.h>
#include <hip/hip_bf16.h>

#define DIM        768
#define NUM_HEADS  12
#define HEAD_DIM   64
#define PATCH_W    64
#define NGROUPS    1024            // 16 * 64
#define MTOT       65536           // NGROUPS * PATCH_W
#define SCALE      0.125f          // 64^-0.5

#define USE_ASYNC_LDS 1            // CDNA5 async global->LDS (ASYNCcnt) staging

typedef __attribute__((ext_vector_type(16))) __bf16 v16bf;
typedef __attribute__((ext_vector_type(8)))  __bf16 v8bf;
typedef __attribute__((ext_vector_type(2)))  __bf16 v2bf;
typedef __attribute__((ext_vector_type(8)))  float  v8f;

// ---------- conversion helpers (round-to-nearest, ties-away) ----------

__device__ __forceinline__ __bf16 f2bf(float f) {
    unsigned u = __builtin_bit_cast(unsigned, f) + 0x8000u;
    unsigned short s = (unsigned short)(u >> 16);
    return __builtin_bit_cast(__bf16, s);
}

// pack two floats into packed bf16x2: 2x v_add + 1x v_perm_b32
__device__ __forceinline__ v2bf pk2bf(float a, float b) {
#if __has_builtin(__builtin_amdgcn_cvt_pk_bf16_f32)
    return __builtin_amdgcn_cvt_pk_bf16_f32(a, b);
#elif __has_builtin(__builtin_amdgcn_perm)
    unsigned ua = __builtin_bit_cast(unsigned, a) + 0x8000u;
    unsigned ub = __builtin_bit_cast(unsigned, b) + 0x8000u;
    // bytes: dst = { ub[3], ub[2], ua[3], ua[2] }  (lo16 = a, hi16 = b)
    unsigned r = __builtin_amdgcn_perm(ub, ua, 0x07060302u);
    return __builtin_bit_cast(v2bf, r);
#else
    unsigned ua = __builtin_bit_cast(unsigned, a) + 0x8000u;
    unsigned ub = __builtin_bit_cast(unsigned, b) + 0x8000u;
    unsigned r = (ub & 0xFFFF0000u) | (ua >> 16);
    return __builtin_bit_cast(v2bf, r);
#endif
}

// ---------- WMMA fragment helpers ----------

// A fragment: 16x32 bf16, row-major source. Lane m = lane&15; half = lane>>4
// holds K = half*8+{0..7} and 16+half*8+{0..7}
__device__ __forceinline__ v16bf load_frag_a(const __bf16* base, int stride, int lane) {
    int m  = lane & 15;
    int kh = (lane >> 4) * 8;
    const v8bf lo = *(const v8bf*)(base + m * stride + kh);
    const v8bf hi = *(const v8bf*)(base + m * stride + kh + 16);
    return __builtin_shufflevector(lo, hi, 0,1,2,3,4,5,6,7,8,9,10,11,12,13,14,15);
}

// B fragment: 32x16 bf16, memory holds B^T row-major. Lane n = lane&15;
// K = (lane>>4)*8 + {0..7} and 16 + (lane>>4)*8 + {0..7}
__device__ __forceinline__ v16bf load_frag_b(const __bf16* base, int stride, int lane) {
    int n  = lane & 15;
    int k0 = (lane >> 4) * 8;
    const v8bf lo = *(const v8bf*)(base + n * stride + k0);
    const v8bf hi = *(const v8bf*)(base + n * stride + k0 + 16);
    return __builtin_shufflevector(lo, hi, 0,1,2,3,4,5,6,7,8,9,10,11,12,13,14,15);
}

__device__ __forceinline__ v8f wmma_bf16(v16bf a, v16bf b, v8f c) {
    return __builtin_amdgcn_wmma_f32_16x16x32_bf16(
        false, a, false, b, (short)0, c, false, false);
}

#if USE_ASYNC_LDS
// 16B async copy global -> LDS (per-lane addresses), tracked by ASYNCcnt.
__device__ __forceinline__ void async_copy16(const void* g, void* l) {
    unsigned           lds = (unsigned)(uintptr_t)l;   // LDS addr = generic addr[31:0]
    unsigned long long ga  = (unsigned long long)(uintptr_t)g;
    asm volatile("global_load_async_to_lds_b128 %0, %1, off"
                 :: "v"(lds), "v"(ga) : "memory");
}
__device__ __forceinline__ void async_wait0() {
    asm volatile("s_wait_asynccnt 0x0" ::: "memory");
}
#endif

// =====================================================================
// Kernel 1: QKV projection  y = x @ w_qkv^T  (M=65536, N=2304, K=768)
// block = 256 threads (8 waves), tile 128(M) x 128(N), K chunk 32,
// register double-buffered staging; scatter-store Q / K / V^T (bf16).
// =====================================================================
__global__ __launch_bounds__(256)
void qkv_gemm_kernel(const float* __restrict__ x,
                     const float* __restrict__ w_qkv,
                     __bf16* __restrict__ Qb,
                     __bf16* __restrict__ Kb,
                     __bf16* __restrict__ Vtb) {
    __shared__ __align__(16) __bf16 As[128 * 32];   // x tile   (M x K)
    __shared__ __align__(16) __bf16 Bs[128 * 32];   // w rows   (N x K) == B^T

    const int tid  = threadIdx.x;
    const int lane = tid & 31;
    const int wv   = tid >> 5;
    const int m0   = blockIdx.x * 128;
    const int n0   = blockIdx.y * 128;
    const int mq   = wv & 3;
    const int ng   = wv >> 2;

    // staging geometry: 512 chunks of 8 elems; thread owns chunk tid and tid+256
    const int r0 = tid >> 2,           c0 = (tid & 3) * 8;
    const int r1 = (tid + 256) >> 2,   c1 = ((tid + 256) & 3) * 8;

    float aR[16], bR[16];
    v8f acc[2][4];
    #pragma unroll
    for (int i = 0; i < 2; ++i)
        #pragma unroll
        for (int j = 0; j < 4; ++j) acc[i][j] = (v8f)0.0f;

    {   // prefetch k-chunk 0
        const float* sA0 = x + (size_t)(m0 + r0) * DIM + c0;
        const float* sA1 = x + (size_t)(m0 + r1) * DIM + c1;
        const float* sB0 = w_qkv + (size_t)(n0 + r0) * DIM + c0;
        const float* sB1 = w_qkv + (size_t)(n0 + r1) * DIM + c1;
        #pragma unroll
        for (int i = 0; i < 8; ++i) {
            aR[i] = sA0[i]; aR[8 + i] = sA1[i];
            bR[i] = sB0[i]; bR[8 + i] = sB1[i];
        }
    }

    for (int kt = 0; kt < DIM; kt += 32) {
        // commit prefetched registers -> LDS (packed fp32 -> bf16)
        #pragma unroll
        for (int i = 0; i < 8; i += 2) {
            *(v2bf*)(As + r0 * 32 + c0 + i) = pk2bf(aR[i],     aR[i + 1]);
            *(v2bf*)(As + r1 * 32 + c1 + i) = pk2bf(aR[8 + i], aR[9 + i]);
            *(v2bf*)(Bs + r0 * 32 + c0 + i) = pk2bf(bR[i],     bR[i + 1]);
            *(v2bf*)(Bs + r1 * 32 + c1 + i) = pk2bf(bR[8 + i], bR[9 + i]);
        }
        __syncthreads();

        // issue next chunk's global loads before the math
        if (kt + 32 < DIM) {
            const float* sA0 = x + (size_t)(m0 + r0) * DIM + kt + 32 + c0;
            const float* sA1 = x + (size_t)(m0 + r1) * DIM + kt + 32 + c1;
            const float* sB0 = w_qkv + (size_t)(n0 + r0) * DIM + kt + 32 + c0;
            const float* sB1 = w_qkv + (size_t)(n0 + r1) * DIM + kt + 32 + c1;
            #pragma unroll
            for (int i = 0; i < 8; ++i) {
                aR[i] = sA0[i]; aR[8 + i] = sA1[i];
                bR[i] = sB0[i]; bR[8 + i] = sB1[i];
            }
        }

        v16bf bf[4];
        #pragma unroll
        for (int j = 0; j < 4; ++j)
            bf[j] = load_frag_b(Bs + (ng * 4 + j) * 16 * 32, 32, lane);
        #pragma unroll
        for (int mi = 0; mi < 2; ++mi) {
            v16bf a = load_frag_a(As + (mq * 2 + mi) * 16 * 32, 32, lane);
            #pragma unroll
            for (int j = 0; j < 4; ++j)
                acc[mi][j] = wmma_bf16(a, bf[j], acc[mi][j]);
        }
        __syncthreads();
    }

    // epilogue: scatter to Q / K / V^T (bf16)
    const int col = lane & 15;
    const int rb  = (lane >> 4) * 8;
    #pragma unroll
    for (int j = 0; j < 4; ++j) {
        int n     = n0 + (ng * 4 + j) * 16 + col;
        int which = (n >= 2 * DIM) ? 2 : ((n >= DIM) ? 1 : 0);  // subtile-uniform
        int rem   = n - which * DIM;
        int head  = rem >> 6;
        int d     = rem & 63;
        #pragma unroll
        for (int mi = 0; mi < 2; ++mi) {
            #pragma unroll
            for (int r = 0; r < 8; ++r) {
                int m  = m0 + (mq * 2 + mi) * 16 + rb + r;
                int g  = m >> 6;
                int wq = m & 63;
                size_t ghoff = (size_t)(g * NUM_HEADS + head) << 12;
                __bf16 v = f2bf(acc[mi][j][r]);
                if (which == 0)      Qb [ghoff + ((size_t)wq << 6) + d] = v;
                else if (which == 1) Kb [ghoff + ((size_t)wq << 6) + d] = v;
                else                 Vtb[ghoff + ((size_t)d  << 6) + wq] = v;
            }
        }
    }
}

// =====================================================================
// Kernel 2: per-(group,head) attention. block = 128 threads (4 waves).
// =====================================================================
__global__ __launch_bounds__(128)
void row_attn_kernel(const __bf16* __restrict__ Qb,
                     const __bf16* __restrict__ Kb,
                     const __bf16* __restrict__ Vtb,
                     __bf16* __restrict__ Ob) {
    __shared__ __align__(16) __bf16 Qs [64 * 64];
    __shared__ __align__(16) __bf16 Ks [64 * 64];
    __shared__ __align__(16) __bf16 Vts[64 * 64];   // V^T : [d][w]
    __shared__ __align__(16) float  Ss [64 * 65];   // scores f32 (padded)
    __shared__ __align__(16) __bf16 Ps [64 * 64];   // probs bf16

    const int tid  = threadIdx.x;
    const int lane = tid & 31;
    const int wv   = tid >> 5;
    const int gh   = blockIdx.x;
    const int g    = gh / NUM_HEADS;
    const int h    = gh - g * NUM_HEADS;
    const size_t base = (size_t)gh << 12;

    // ---- stage Q, K, V^T into LDS ----
#if USE_ASYNC_LDS
    {
        const uint4* gq = (const uint4*)(Qb  + base);
        const uint4* gk = (const uint4*)(Kb  + base);
        const uint4* gv = (const uint4*)(Vtb + base);
        uint4* sq = (uint4*)Qs; uint4* sk = (uint4*)Ks; uint4* sv = (uint4*)Vts;
        for (int i = tid; i < 512; i += 128) {
            async_copy16(gq + i, sq + i);
            async_copy16(gk + i, sk + i);
            async_copy16(gv + i, sv + i);
        }
        async_wait0();
    }
#else
    {
        const uint4* gq = (const uint4*)(Qb  + base);
        const uint4* gk = (const uint4*)(Kb  + base);
        const uint4* gv = (const uint4*)(Vtb + base);
        uint4* sq = (uint4*)Qs; uint4* sk = (uint4*)Ks; uint4* sv = (uint4*)Vts;
        for (int i = tid; i < 512; i += 128) { sq[i] = gq[i]; sk[i] = gk[i]; sv[i] = gv[i]; }
    }
#endif
    __syncthreads();

    const int m0  = wv * 16;
    const int col = lane & 15;
    const int rb  = (lane >> 4) * 8;

    // ---- S = Q K^T * scale ----
    #pragma unroll
    for (int nt = 0; nt < 4; ++nt) {
        v8f acc = (v8f)0.0f;
        #pragma unroll
        for (int kk = 0; kk < HEAD_DIM; kk += 32) {
            v16bf a = load_frag_a(Qs + m0 * 64 + kk, 64, lane);
            v16bf b = load_frag_b(Ks + nt * 16 * 64 + kk, 64, lane);
            acc = wmma_bf16(a, b, acc);
        }
        #pragma unroll
        for (int r = 0; r < 8; ++r)
            Ss[(m0 + rb + r) * 65 + nt * 16 + col] = acc[r] * SCALE;
    }
    __syncthreads();

    // ---- row softmax (threads 0..63, one row each); recompute exp ----
    if (tid < 64) {
        const float* srow = Ss + tid * 65;
        float mx = srow[0];
        #pragma unroll 8
        for (int c = 1; c < 64; ++c) mx = fmaxf(mx, srow[c]);
        float sum = 0.0f;
        #pragma unroll 8
        for (int c = 0; c < 64; ++c) sum += __expf(srow[c] - mx);
        float inv = 1.0f / sum;
        __bf16* prow = Ps + tid * 64;
        #pragma unroll 8
        for (int c = 0; c < 64; c += 2)
            *(v2bf*)(prow + c) = pk2bf(__expf(srow[c] - mx) * inv,
                                       __expf(srow[c + 1] - mx) * inv);
    }
    __syncthreads();

    // ---- O = P V  (B columns = rows of V^T) ----
    #pragma unroll
    for (int nt = 0; nt < 4; ++nt) {
        v8f acc = (v8f)0.0f;
        #pragma unroll
        for (int kk = 0; kk < PATCH_W; kk += 32) {
            v16bf a = load_frag_a(Ps + m0 * 64 + kk, 64, lane);
            v16bf b = load_frag_b(Vts + nt * 16 * 64 + kk, 64, lane);
            acc = wmma_bf16(a, b, acc);
        }
        #pragma unroll
        for (int r = 0; r < 8; ++r) {
            int q = m0 + rb + r;
            int d = nt * 16 + col;
            Ob[((size_t)(g * 64 + q)) * DIM + h * HEAD_DIM + d] = f2bf(acc[r]);
        }
    }
}

// =====================================================================
// Kernel 3: out = O @ w_proj^T + b  (M=65536, N=768, K=768), f32 out.
// =====================================================================
__global__ __launch_bounds__(256)
void proj_gemm_kernel(const __bf16* __restrict__ Ob,
                      const float* __restrict__ w_proj,
                      const float* __restrict__ b_proj,
                      float* __restrict__ out) {
    __shared__ __align__(16) __bf16 As[128 * 32];
    __shared__ __align__(16) __bf16 Bs[128 * 32];

    const int tid  = threadIdx.x;
    const int lane = tid & 31;
    const int wv   = tid >> 5;
    const int m0   = blockIdx.x * 128;
    const int n0   = blockIdx.y * 128;
    const int mq   = wv & 3;
    const int ng   = wv >> 2;

    const int r0 = tid >> 2,           c0 = (tid & 3) * 8;
    const int r1 = (tid + 256) >> 2,   c1 = ((tid + 256) & 3) * 8;

    v8bf aR[2];
    float bR[16];
    v8f acc[2][4];
    #pragma unroll
    for (int i = 0; i < 2; ++i)
        #pragma unroll
        for (int j = 0; j < 4; ++j) acc[i][j] = (v8f)0.0f;

    {
        aR[0] = *(const v8bf*)(Ob + (size_t)(m0 + r0) * DIM + c0);
        aR[1] = *(const v8bf*)(Ob + (size_t)(m0 + r1) * DIM + c1);
        const float* sB0 = w_proj + (size_t)(n0 + r0) * DIM + c0;
        const float* sB1 = w_proj + (size_t)(n0 + r1) * DIM + c1;
        #pragma unroll
        for (int i = 0; i < 8; ++i) { bR[i] = sB0[i]; bR[8 + i] = sB1[i]; }
    }

    for (int kt = 0; kt < DIM; kt += 32) {
        *(v8bf*)(As + r0 * 32 + c0) = aR[0];
        *(v8bf*)(As + r1 * 32 + c1) = aR[1];
        #pragma unroll
        for (int i = 0; i < 8; i += 2) {
            *(v2bf*)(Bs + r0 * 32 + c0 + i) = pk2bf(bR[i],     bR[i + 1]);
            *(v2bf*)(Bs + r1 * 32 + c1 + i) = pk2bf(bR[8 + i], bR[9 + i]);
        }
        __syncthreads();

        if (kt + 32 < DIM) {
            aR[0] = *(const v8bf*)(Ob + (size_t)(m0 + r0) * DIM + kt + 32 + c0);
            aR[1] = *(const v8bf*)(Ob + (size_t)(m0 + r1) * DIM + kt + 32 + c1);
            const float* sB0 = w_proj + (size_t)(n0 + r0) * DIM + kt + 32 + c0;
            const float* sB1 = w_proj + (size_t)(n0 + r1) * DIM + kt + 32 + c1;
            #pragma unroll
            for (int i = 0; i < 8; ++i) { bR[i] = sB0[i]; bR[8 + i] = sB1[i]; }
        }

        v16bf bf[4];
        #pragma unroll
        for (int j = 0; j < 4; ++j)
            bf[j] = load_frag_b(Bs + (ng * 4 + j) * 16 * 32, 32, lane);
        #pragma unroll
        for (int mi = 0; mi < 2; ++mi) {
            v16bf a = load_frag_a(As + (mq * 2 + mi) * 16 * 32, 32, lane);
            #pragma unroll
            for (int j = 0; j < 4; ++j)
                acc[mi][j] = wmma_bf16(a, bf[j], acc[mi][j]);
        }
        __syncthreads();
    }

    const int col = lane & 15;
    const int rb  = (lane >> 4) * 8;
    #pragma unroll
    for (int j = 0; j < 4; ++j) {
        int n = n0 + (ng * 4 + j) * 16 + col;
        float bias = b_proj[n];
        #pragma unroll
        for (int mi = 0; mi < 2; ++mi) {
            #pragma unroll
            for (int r = 0; r < 8; ++r) {
                int m = m0 + (mq * 2 + mi) * 16 + rb + r;
                out[(size_t)m * DIM + n] = acc[mi][j][r] + bias;
            }
        }
    }
}

// =====================================================================
extern "C" void kernel_launch(void* const* d_in, const int* in_sizes, int n_in,
                              void* d_out, int out_size, void* d_ws, size_t ws_size,
                              hipStream_t stream) {
    const float* x      = (const float*)d_in[0];   // (16, 4096, 768)
    const float* w_qkv  = (const float*)d_in[1];   // (2304, 768)
    const float* w_proj = (const float*)d_in[2];   // (768, 768)
    const float* b_proj = (const float*)d_in[3];   // (768,)
    float* out = (float*)d_out;

    const size_t nmat = (size_t)NGROUPS * NUM_HEADS * 64 * 64;
    __bf16* Qb  = (__bf16*)d_ws;
    __bf16* Kb  = Qb  + nmat;
    __bf16* Vtb = Kb  + nmat;
    __bf16* Ob  = Vtb + nmat;

    qkv_gemm_kernel<<<dim3(MTOT / 128, (3 * DIM) / 128), 256, 0, stream>>>(
        x, w_qkv, Qb, Kb, Vtb);

    row_attn_kernel<<<NGROUPS * NUM_HEADS, 128, 0, stream>>>(Qb, Kb, Vtb, Ob);

    proj_gemm_kernel<<<dim3(MTOT / 128, DIM / 128), 256, 0, stream>>>(
        Ob, w_proj, b_proj, out);
}
